// AnatomyNet_35785667510727
// MI455X (gfx1250) — compile-verified, matrix operand order checked by reading, not run on hardware
//
#include <hip/hip_runtime.h>

// ---------------------------------------------------------------------------
// AnatomyNet-style pipeline for MI455X (gfx1250, wave32, WMMA, TDM).
//   conv blocks  -> implicit GEMM, v_wmma_f32_16x16x32_bf16
//                   A: channels-last bf16 activations (32B/ fragment loads)
//                   B: pre-packed bf16 weight fragments, TDM-staged into LDS
//   ROI pooling  -> split-K WMMA GEMM, bf16 NCDHW embedding + f32 atlas
//   expert MLPs  -> per-wave VALU GEMVs (negligible FLOPs)
// ---------------------------------------------------------------------------

typedef __attribute__((ext_vector_type(16))) __bf16        v16bf;
typedef __attribute__((ext_vector_type(8)))  float         v8f;
typedef __attribute__((ext_vector_type(8)))  unsigned int  v8u;
typedef __attribute__((ext_vector_type(4)))  unsigned int  u32x4;
typedef __attribute__((ext_vector_type(8)))  int           i32x8;
typedef __attribute__((ext_vector_type(4)))  int           i32x4;

#define B_   2
#define C_   32
#define D_   96
#define V_   (96 * 96 * 96)      // 884736
#define NR_  100
#define H1_  64
#define RH_  256
#define RE_  128
#define EPS_ 1e-5f
#define WPK_ (27 * 2 * 32 * 8)   // packed weight dwords per layer = 13824

// round-to-nearest-even f32 -> bf16 bits
__device__ __forceinline__ unsigned int bf16_bits(float f) {
  unsigned int u = __builtin_bit_cast(unsigned int, f);
  return (u + 0x7FFFu + ((u >> 16) & 1u)) >> 16;
}

// ---------------------------------------------------------------------------
// conv0: 1 -> 32 channels, 3x3x3 SAME, + bias.  K is only 27, scalar VALU.
// ---------------------------------------------------------------------------
__global__ __launch_bounds__(256) void conv0_kernel(
    const float* __restrict__ in,    // [B][V]
    const float* __restrict__ w,     // [32][27]
    const float* __restrict__ bias,  // [32]
    float* __restrict__ out)         // [B][32][V]
{
  const int idx = blockIdx.x * 256 + threadIdx.x;   // voxel over B*V (exact)
  const int b = idx / V_;
  const int v = idx - b * V_;
  const int z = v / (D_ * D_);
  const int y = (v / D_) % D_;
  const int x = v % D_;

  float patch[27];
#pragma unroll
  for (int tap = 0; tap < 27; ++tap) {
    const int dz = tap / 9 - 1, dy = (tap / 3) % 3 - 1, dx = tap % 3 - 1;
    const int zz = z + dz, yy = y + dy, xx = x + dx;
    const bool ok = (unsigned)zz < (unsigned)D_ && (unsigned)yy < (unsigned)D_ &&
                    (unsigned)xx < (unsigned)D_;
    patch[tap] = ok ? in[(size_t)b * V_ + (size_t)zz * (D_ * D_) + yy * D_ + xx] : 0.f;
  }
  for (int co = 0; co < C_; ++co) {
    float acc = bias[co];
#pragma unroll
    for (int tap = 0; tap < 27; ++tap) acc += w[co * 27 + tap] * patch[tap];
    out[((size_t)b * C_ + co) * V_ + v] = acc;
  }
}

// ---------------------------------------------------------------------------
// Pre-pack conv weights into per-lane WMMA B-fragment images (bf16 pairs):
//   wpk[layer][tap][frag][lane][pair]   (v8u per (tap,frag,lane))
//   lane L: col co = frag*16 + L%16 ; element e -> ci = (L>=16)*16 + e
// ---------------------------------------------------------------------------
__global__ __launch_bounds__(256) void pack_weights_kernel(
    const float* __restrict__ ckw,   // [3][32co][32ci][27]
    unsigned int* __restrict__ wpk)  // [3][WPK_]
{
  const int i = blockIdx.x * 256 + threadIdx.x;    // 0..3*WPK_-1 exact
  const int layer = i / WPK_;
  int r = i - layer * WPK_;
  const int tap  = r >> 9;  r &= 511;
  const int frag = r >> 8;  r &= 255;
  const int lane = r >> 3;
  const int pair = r & 7;
  const int co = frag * 16 + (lane & 15);
  const int ci = (lane >> 4) * 16 + pair * 2;
  const float* w = ckw + (size_t)layer * C_ * C_ * 27;
  const unsigned lo = bf16_bits(w[(co * C_ + ci) * 27 + tap]);
  const unsigned hi = bf16_bits(w[(co * C_ + ci + 1) * 27 + tap]);
  wpk[i] = lo | (hi << 16);
}

// ---------------------------------------------------------------------------
// 32->32 conv as implicit GEMM.  One wave per tile of 16 x-voxels x 32 co.
//   A (16x32 bf16): channels-last activations -> one 32B load per lane/tap
//   B (32x16 bf16): pre-packed fragments in LDS -> ds_load_b128 pairs
// LDS weight image staged by the Tensor Data Mover when available.
// ---------------------------------------------------------------------------
__global__ __launch_bounds__(256) void conv_wmma_kernel(
    const unsigned short* __restrict__ in,   // [B][V][32] bf16 channels-last
    const unsigned int* __restrict__ wsrc,   // [WPK_] packed fragments
    const float* __restrict__ bias,          // [32]
    float* __restrict__ out)                 // [B][32][V] f32
{
  __shared__ v8u ldswv[27 * 2 * 32];         // 55,296 B packed B-fragments
  const int tid = threadIdx.x;

#if defined(__has_builtin) && __has_builtin(__builtin_amdgcn_tensor_load_to_lds) && \
    __has_builtin(__builtin_amdgcn_s_wait_tensorcnt)
  if ((tid >> 5) == 0) {                     // one wave issues the TDM DMA
    const unsigned long long ga = (unsigned long long)(size_t)wsrc;
    const unsigned lds = (unsigned)(size_t)&ldswv[0];
    const u32x4 g0 = {
        1u,                                            // count=1 user D#
        lds,                                           // lds_addr
        (unsigned)ga,                                  // global_addr[31:0]
        (unsigned)((ga >> 32) & 0x1FFFFFFu) | (2u << 30) };  // [56:32] | type=2
    const i32x8 g1 = {
        (int)(2u << 16),             // workgroup_mask=0, data_size=4B
        (int)((WPK_ & 0xFFFFu) << 16),   // tensor_dim0[15:0]
        (int)(1u << 16),             // tensor_dim0[31:16]=0 | tensor_dim1=1
        (int)(WPK_ << 16),           // tensor_dim1 hi=0 | tile_dim0=WPK_
        (int)1,                      // tile_dim1=1 | tile_dim2=0
        (int)WPK_,                   // tensor_dim0_stride[31:0]
        0, 0 };
    const i32x4 gz = {0, 0, 0, 0};
#if __clang_major__ >= 23
    const i32x8 gz8 = {0, 0, 0, 0, 0, 0, 0, 0};
    __builtin_amdgcn_tensor_load_to_lds(g0, g1, gz, gz, gz8, 0);
#else
    __builtin_amdgcn_tensor_load_to_lds(g0, g1, gz, gz, 0);
#endif
    __builtin_amdgcn_s_wait_tensorcnt(0);
  }
#else
  {
    unsigned int* lw = (unsigned int*)ldswv;
    for (int i = tid; i < WPK_; i += 256) lw[i] = wsrc[i];
  }
#endif
  __syncthreads();

  const int lane = tid & 31;
  const int m  = lane & 15;     // A row (x offset) / B col (co) / C col index
  const int hh = lane >> 4;     // K half: 0 -> K 0..15, 1 -> K 16..31

  int t = blockIdx.x * 8 + (tid >> 5);            // tile id, exact 110592
  const int xt  = t % (D_ / 16); t /= (D_ / 16);
  const int yy0 = t % D_;        t /= D_;
  const int zz0 = t % D_;        t /= D_;
  const int b   = t;            // 0..1
  const int x0  = xt * 16;

  const v8f zero = {};
  v8f acc0 = zero, acc1 = zero;

  for (int tap = 0; tap < 27; ++tap) {
    const int dz = tap / 9 - 1, dy = (tap / 3) % 3 - 1, dx = tap % 3 - 1;
    const int iz = zz0 + dz, iy = yy0 + dy;
    const int xx = x0 + m + dx;
    const bool ok = (unsigned)iz < (unsigned)D_ && (unsigned)iy < (unsigned)D_ &&
                    (unsigned)xx < (unsigned)D_;

    // A fragment: 16 consecutive bf16 channels of one voxel (32B aligned)
    v8u ap = {};
    if (ok) {
      const size_t vox = (size_t)b * V_ + (size_t)iz * (D_ * D_) + iy * D_ + xx;
      ap = *(const v8u*)(in + vox * C_ + hh * 16);
    }
    const v16bf a = __builtin_bit_cast(v16bf, ap);

    // B fragments: contiguous per-lane images in LDS (2x ds_load_b128 each)
    const v8u b0p = ldswv[(tap * 2 + 0) * 32 + lane];
    const v8u b1p = ldswv[(tap * 2 + 1) * 32 + lane];
    const v16bf b0 = __builtin_bit_cast(v16bf, b0p);
    const v16bf b1 = __builtin_bit_cast(v16bf, b1p);

    acc0 = __builtin_amdgcn_wmma_f32_16x16x32_bf16(false, a, false, b0,
                                                   (short)0, acc0, false, false);
    acc1 = __builtin_amdgcn_wmma_f32_16x16x32_bf16(false, a, false, b1,
                                                   (short)0, acc1, false, false);
  }

  // C layout: lane holds col N=lane%16, VGPR j holds row M = j + 8*(lane>=16)
  const float bi0 = bias[m];
  const float bi1 = bias[m + 16];
  const size_t base0 = ((size_t)b * C_ + m) * V_ +
                       (size_t)zz0 * (D_ * D_) + yy0 * D_ + x0 + 8 * hh;
  const size_t base1 = base0 + (size_t)16 * V_;
#pragma unroll
  for (int j = 0; j < 8; ++j) {
    out[base0 + j] = acc0[j] + bi0;
    out[base1 + j] = acc1[j] + bi1;
  }
}

// ---------------------------------------------------------------------------
// Instance-norm statistics: one block per (b,c), reduce V elements.
// ---------------------------------------------------------------------------
__global__ __launch_bounds__(256) void stats_kernel(
    const float* __restrict__ y, float2* __restrict__ mr)
{
  __shared__ float ssum[256], ssq[256];
  const int bc = blockIdx.x;                  // 0..63
  const float* p = y + (size_t)bc * V_;
  float s = 0.f, q = 0.f;
  for (int i = threadIdx.x; i < V_; i += 256) {
    const float v = p[i];
    s += v; q += v * v;
  }
  ssum[threadIdx.x] = s; ssq[threadIdx.x] = q;
  __syncthreads();
  for (int off = 128; off > 0; off >>= 1) {
    if (threadIdx.x < off) {
      ssum[threadIdx.x] += ssum[threadIdx.x + off];
      ssq[threadIdx.x]  += ssq[threadIdx.x + off];
    }
    __syncthreads();
  }
  if (threadIdx.x == 0) {
    const float mean = ssum[0] * (1.f / (float)V_);
    const float var  = ssq[0] * (1.f / (float)V_) - mean * mean;
    mr[bc] = make_float2(mean, rsqrtf(var + EPS_));
  }
}

// ---------------------------------------------------------------------------
// normalize + ReLU, NCDHW f32 -> channels-last bf16 [b][v][32]
// (next conv layer's A operand; one 64B store per voxel)
// ---------------------------------------------------------------------------
__global__ __launch_bounds__(256) void norm_nhwc_bf16_kernel(
    const float* __restrict__ y, const float2* __restrict__ mr,
    uint4* __restrict__ outb)
{
  const int idx = blockIdx.x * 256 + threadIdx.x;   // b*V + v (exact)
  const int b = idx / V_;
  const int v = idx - b * V_;
  unsigned int d[16];
#pragma unroll
  for (int c = 0; c < C_; c += 2) {
    const float2 m0 = mr[b * C_ + c];
    const float2 m1 = mr[b * C_ + c + 1];
    const float f0 = fmaxf((y[((size_t)b * C_ + c)     * V_ + v] - m0.x) * m0.y, 0.f);
    const float f1 = fmaxf((y[((size_t)b * C_ + c + 1) * V_ + v] - m1.x) * m1.y, 0.f);
    d[c >> 1] = bf16_bits(f0) | (bf16_bits(f1) << 16);
  }
  uint4* o = outb + (size_t)idx * 4;
#pragma unroll
  for (int k = 0; k < 4; ++k)
    o[k] = make_uint4(d[4 * k], d[4 * k + 1], d[4 * k + 2], d[4 * k + 3]);
}

// ---------------------------------------------------------------------------
// normalize + ReLU, NCDHW f32 -> NCDHW bf16  (ROI GEMM's B operand)
// ---------------------------------------------------------------------------
__global__ __launch_bounds__(256) void norm_nchw_bf16_kernel(
    const float4* __restrict__ y, const float2* __restrict__ mr,
    uint4* __restrict__ outb)
{
  const size_t i  = (size_t)blockIdx.x * 256 + threadIdx.x;  // 8 floats each
  const int    bc = (int)((i * 8) / (size_t)V_);
  const float2 m  = mr[bc];
  const float4 a = y[i * 2], b = y[i * 2 + 1];
  const float r0 = fmaxf((a.x - m.x) * m.y, 0.f);
  const float r1 = fmaxf((a.y - m.x) * m.y, 0.f);
  const float r2 = fmaxf((a.z - m.x) * m.y, 0.f);
  const float r3 = fmaxf((a.w - m.x) * m.y, 0.f);
  const float r4 = fmaxf((b.x - m.x) * m.y, 0.f);
  const float r5 = fmaxf((b.y - m.x) * m.y, 0.f);
  const float r6 = fmaxf((b.z - m.x) * m.y, 0.f);
  const float r7 = fmaxf((b.w - m.x) * m.y, 0.f);
  outb[i] = make_uint4(bf16_bits(r0) | (bf16_bits(r1) << 16),
                       bf16_bits(r2) | (bf16_bits(r3) << 16),
                       bf16_bits(r4) | (bf16_bits(r5) << 16),
                       bf16_bits(r6) | (bf16_bits(r7) << 16));
}

// ---------------------------------------------------------------------------
// ROI pooling as split-K WMMA GEMM:
//   D[r, bc] = sum_v mask[r,v] * emb[bc,v],  M=112 (100 padded), N=64, K=V
// Wave = (mtile, kgroup); 4 N-tile accumulators; atomic reduction.
// Mask row-sums folded into the A loads (each (r,v) touched exactly once).
// ---------------------------------------------------------------------------
#define KG_ 4096                      // K elements per kgroup; 216 groups exact
__global__ __launch_bounds__(256) void roi_wmma_kernel(
    const unsigned short* __restrict__ emb,   // [64][V] bf16
    const float* __restrict__ mask,           // [100][V] f32
    float* __restrict__ roi_acc,              // [B][NR][C]
    float* __restrict__ masksum)              // [NR]
{
  const int tid  = threadIdx.x;
  const int lane = tid & 31;
  const int m  = lane & 15;
  const int hh = lane >> 4;
  const int wid = blockIdx.x * 8 + (tid >> 5);   // 0..1511 exact
  const int mt = wid % 7;
  const int kg = wid / 7;
  const int rbase = mt * 16;
  const int r = rbase + m;
  const bool rok = r < NR_;
  const size_t v0 = (size_t)kg * KG_;

  const v8f zero = {};
  v8f acc[4] = {zero, zero, zero, zero};
  float msum = 0.f;

  for (int c = 0; c < KG_; c += 32) {
    const size_t vk = v0 + c + hh * 16;

    // A fragment: 16 consecutive mask f32 of row r -> bf16 (+ row sum)
    v8u ap = {};
    if (rok) {
      const float* p = mask + (size_t)r * V_ + vk;
      __builtin_prefetch(p + 64, 0, 0);          // global_prefetch_b8
#pragma unroll
      for (int e = 0; e < 16; e += 2) {
        const float f0 = p[e], f1 = p[e + 1];
        msum += f0 + f1;
        ap[e >> 1] = bf16_bits(f0) | (bf16_bits(f1) << 16);
      }
    }
    const v16bf a = __builtin_bit_cast(v16bf, ap);

#pragma unroll
    for (int nt = 0; nt < 4; ++nt) {
      const int bc = nt * 16 + m;                 // (b,c) column for this lane
      const v8u bp = *(const v8u*)(emb + (size_t)bc * V_ + vk);  // 32B bf16
      const v16bf bb = __builtin_bit_cast(v16bf, bp);
      acc[nt] = __builtin_amdgcn_wmma_f32_16x16x32_bf16(false, a, false, bb,
                                                        (short)0, acc[nt],
                                                        false, false);
    }
  }

  if (rok) atomicAdd(&masksum[r], msum);          // disjoint K halves per lane

  // C layout: lane col = lane%16 within N-tile; rows rr = rbase + j + 8*hh
#pragma unroll
  for (int nt = 0; nt < 4; ++nt) {
    const int bc = nt * 16 + m;
    const int b = bc >> 5, cch = bc & 31;
#pragma unroll
    for (int j = 0; j < 8; ++j) {
      const int rr = rbase + j + 8 * hh;
      if (rr < NR_)
        atomicAdd(&roi_acc[((size_t)b * NR_ + rr) * C_ + cch], acc[nt][j]);
    }
  }
}

// ---------------------------------------------------------------------------
// Per-(b,r) expert MLP chain.  One wave per (b,r); 4 waves / block.
// ---------------------------------------------------------------------------
__global__ __launch_bounds__(128) void mlp_kernel(
    const float* __restrict__ roi_acc, const float* __restrict__ masksum,
    const float* __restrict__ sw1, const float* __restrict__ sb1,
    const float* __restrict__ sw2, const float* __restrict__ sb2,
    const float* __restrict__ pw1, const float* __restrict__ pb1,
    const float* __restrict__ pw2, const float* __restrict__ pb2,
    float* __restrict__ out)
{
  __shared__ float rc [4][C_];
  __shared__ float h1 [4][H1_];
  __shared__ float sfv[4][C_];
  __shared__ float h2 [4][RH_];

  const int wv   = threadIdx.x >> 5;
  const int lane = threadIdx.x & 31;
  const int id   = blockIdx.x * 4 + wv;           // 0..199 exact (50 blocks)
  const int r = id % NR_;
  const int b = id / NR_;

  const float rcv = roi_acc[((size_t)b * NR_ + r) * C_ + lane] / masksum[r];
  rc[wv][lane] = rcv;
  __syncthreads();

  const float* W1 = sw1 + (size_t)r * C_ * H1_;
#pragma unroll
  for (int t = 0; t < 2; ++t) {
    const int h = lane + 32 * t;
    float a = sb1[r * H1_ + h];
    for (int c = 0; c < C_; ++c) a += rc[wv][c] * W1[c * H1_ + h];
    h1[wv][h] = fmaxf(a, 0.f);
  }
  __syncthreads();

  const float* W2 = sw2 + (size_t)r * H1_ * C_;
  float a2 = sb2[r * C_ + lane];
  for (int h = 0; h < H1_; ++h) a2 += h1[wv][h] * W2[h * C_ + lane];
  const float sc = 1.f / (1.f + __expf(-a2));
  sfv[wv][lane] = sc * rcv;
  __syncthreads();

  const float* P1 = pw1 + (size_t)r * C_ * RH_;
#pragma unroll
  for (int t = 0; t < 8; ++t) {
    const int h = lane + 32 * t;
    float a = pb1[r * RH_ + h];
    for (int c = 0; c < C_; ++c) a += sfv[wv][c] * P1[c * RH_ + h];
    h2[wv][h] = fmaxf(a, 0.f);
  }
  __syncthreads();

  const float* P2 = pw2 + (size_t)r * RH_ * RE_;
#pragma unroll
  for (int t = 0; t < 4; ++t) {
    const int e = lane + 32 * t;
    float a = pb2[r * RE_ + e];
    for (int h = 0; h < RH_; ++h) a += h2[wv][h] * P2[h * RE_ + e];
    out[((size_t)b * NR_ + r) * RE_ + e] = a;
  }
}

// ---------------------------------------------------------------------------
extern "C" void kernel_launch(void* const* d_in, const int* in_sizes, int n_in,
                              void* d_out, int out_size, void* d_ws, size_t ws_size,
                              hipStream_t stream) {
  (void)in_sizes; (void)n_in; (void)out_size; (void)ws_size;

  const float* data  = (const float*)d_in[0];
  const float* atlas = (const float*)d_in[1];
  const float* c0w   = (const float*)d_in[2];
  const float* c0b   = (const float*)d_in[3];
  const float* ckw   = (const float*)d_in[4];   // [3][32][32][27]
  const float* ckb   = (const float*)d_in[5];   // [3][32]
  const float* sw1   = (const float*)d_in[6];
  const float* sb1   = (const float*)d_in[7];
  const float* sw2   = (const float*)d_in[8];
  const float* sb2   = (const float*)d_in[9];
  const float* pw1   = (const float*)d_in[10];
  const float* pb1   = (const float*)d_in[11];
  const float* pw2   = (const float*)d_in[12];
  const float* pb2   = (const float*)d_in[13];
  float* out = (float*)d_out;

  char* ws = (char*)d_ws;
  const size_t ybytes = (size_t)B_ * C_ * V_ * sizeof(float);           // 226,492,416
  const size_t ebytes = (size_t)B_ * C_ * V_ * sizeof(unsigned short);  // 113,246,208
  float*          bufY = (float*)(ws);
  unsigned short* embb = (unsigned short*)(ws + ybytes);   // bf16, both layouts
  unsigned int*   wpk  = (unsigned int*)(ws + ybytes + ebytes);
  char*           tail = ws + ybytes + ebytes + (size_t)3 * WPK_ * 4;
  float2*         mr   = (float2*)tail;                    // 64 float2
  float*          roi  = (float*)(tail + 1024);            // 6400 f32
  float*          msum = roi + B_ * NR_ * C_;              // 100 f32

  // zero split-K accumulators every call (graph-capture safe)
  hipMemsetAsync(roi, 0, (size_t)(B_ * NR_ * C_ + NR_) * sizeof(float), stream);

  // pack conv weights into WMMA B-fragment images (3 layers)
  pack_weights_kernel<<<(3 * WPK_) / 256, 256, 0, stream>>>(ckw, wpk);

  // conv0 + inorm + relu -> channels-last bf16
  conv0_kernel<<<(B_ * V_) / 256, 256, 0, stream>>>(data, c0w, c0b, bufY);
  stats_kernel<<<B_ * C_, 256, 0, stream>>>(bufY, mr);
  norm_nhwc_bf16_kernel<<<(B_ * V_) / 256, 256, 0, stream>>>(bufY, mr, (uint4*)embb);

  // three WMMA conv blocks
  for (int i = 0; i < 3; ++i) {
    conv_wmma_kernel<<<(B_ * D_ * D_ * (D_ / 16)) / 8, 256, 0, stream>>>(
        embb, wpk + (size_t)i * WPK_, ckb + i * C_, bufY);
    stats_kernel<<<B_ * C_, 256, 0, stream>>>(bufY, mr);
    if (i < 2)
      norm_nhwc_bf16_kernel<<<(B_ * V_) / 256, 256, 0, stream>>>(bufY, mr, (uint4*)embb);
    else  // final layer feeds ROI pooling: NCDHW bf16
      norm_nchw_bf16_kernel<<<(B_ * C_ * V_) / 2048, 256, 0, stream>>>(
          (const float4*)bufY, mr, (uint4*)embb);
  }

  // split-K WMMA ROI pooling: 7 mtiles x 216 kgroups = 1512 waves = 189 blocks
  roi_wmma_kernel<<<(7 * (V_ / KG_)) / 8, 256, 0, stream>>>(embb, atlas, roi, msum);

  // per-ROI MLPs -> output [2,100,128]
  mlp_kernel<<<(B_ * NR_) / 4, 128, 0, stream>>>(
      roi, msum, sw1, sb1, sw2, sb2, pw1, pb1, pw2, pb2, out);
}
